// RAFT_7138235646476
// MI455X (gfx1250) — matrix operand order repaired and checked by reading, not compiled, not run on hardware
//
#include <hip/hip_runtime.h>
#include <stdint.h>

// ===========================================================================
// RAFT encoder + all-pairs correlation + pyramid on MI455X (gfx1250, wave32).
//  - Convs: implicit GEMM, 32Cout x 256px block tile, K chunked by 64,
//    8x v_wmma_f32_16x16x32_bf16 per wave per chunk, LDS staged with packed
//    64-bit DS stores, fragments read as contiguous 32/128-bit DS loads.
//  - Correlation: TDM (tensor_load_to_lds, 6-arg clang-23 form) double-buffered
//    fp32 tile staging, synchronized with s_wait_tensorcnt + barrier; WMMA
//    bf16 compute.
//  - Roofline: convs ~150 GFLOP (compute-limited -> matrix pipe);
//    correlation writes 170MB (HBM-bound at 23.3 TB/s -> bf16 inputs free).
// ===========================================================================

typedef __bf16 bf16;
typedef __attribute__((ext_vector_type(16))) __bf16 v16bf;
typedef __attribute__((ext_vector_type(4)))  __bf16 v4bf;
typedef __attribute__((ext_vector_type(8)))  float  v8f;
typedef __attribute__((ext_vector_type(4)))  float  f32x4;

#define INORM_EPS 1e-5f

#if defined(__HIP_DEVICE_COMPILE__) && __has_builtin(__builtin_amdgcn_tensor_load_to_lds) && \
    __has_builtin(__builtin_amdgcn_s_wait_tensorcnt)
#define USE_TDM 1
typedef unsigned int uint32x4 __attribute__((ext_vector_type(4)));
typedef int          int32x4  __attribute__((ext_vector_type(4)));
typedef int          int32x8  __attribute__((ext_vector_type(8)));
#else
#define USE_TDM 0
#endif

union FragBF {
  v16bf    v;
  uint32_t u[8];
};

// ---------------------------------------------------------------------------
// Fragment loaders for v_wmma_f32_16x16x32_bf16 (A/B k order contiguous; C/D per ISA 7.12.2):
//  A 16x32: lanes 0-15 M=lane; VGPR0..3 hold K 0..7 (+16 for lanes 16-31),
//           VGPR4..7 hold K 8..15 (+16).
//  B 32x16: lanes 0-15 K 0..15 (2/VGPR), lanes 16-31 K 16..31; N = lane%16.
//  C/D 16x16 f32: VGPR v -> M = v + 8*(lane>=16), N = lane%16.
// LDS is stored K-contiguous so fragment loads are wide DS reads.
// ---------------------------------------------------------------------------

// sA: bf16 [32 m][64 k] row-major. Returns A frag for rows m0..m0+15, K sub-
// chunk kc (0/1 -> K 0..31 / 32..63). Two 16B-contiguous LDS reads per lane.
__device__ inline v16bf frag_a_bf16(const bf16* sA, int m0, int kc, int lane) {
  const int half = lane >> 4, row = lane & 15;
  const uint32_t* r =
      (const uint32_t*)(sA + (size_t)(m0 + row) * 64 + kc * 32);
  FragBF f;
#pragma unroll
  for (int v = 0; v < 4; ++v) f.u[v] = r[half * 8 + v];
#pragma unroll
  for (int v = 0; v < 4; ++v) f.u[4 + v] = r[half * 8 + 4 + v];
  return f.v;
}

// sB: bf16 [px][64 k] row-major. Returns B frag for columns col0..col0+15,
// K sub-chunk kc. One 32B-contiguous LDS read region per lane.
__device__ inline v16bf frag_b_bf16(const bf16* sB, int col0, int kc, int lane) {
  const int half = lane >> 4, n = lane & 15;
  const uint32_t* q =
      (const uint32_t*)(sB + (size_t)(col0 + n) * 64 + kc * 32 + half * 16);
  FragBF f;
#pragma unroll
  for (int v = 0; v < 8; ++v) f.u[v] = q[v];
  return f.v;
}

// fp32 LDS variants (TDM stages raw fp32 tiles): layout [k][cols], convert on
// fragment load.
__device__ inline v16bf frag_a_f32(const float* sA, int ldc, int m0, int kc,
                                   int lane) {
  const int half = lane >> 4, row = lane & 15;
  v16bf a;
#pragma unroll
  for (int v = 0; v < 8; ++v) {
#pragma unroll
    for (int s = 0; s < 2; ++s) {
      const int k = kc * 32 + half * 16 + v * 2 + s;
      a[v * 2 + s] = (bf16)sA[(size_t)k * ldc + m0 + row];
    }
  }
  return a;
}

__device__ inline v16bf frag_b_f32(const float* sB, int ldc, int col0, int kc,
                                   int lane) {
  const int half = lane >> 4, n = lane & 15;
  v16bf b;
#pragma unroll
  for (int v = 0; v < 8; ++v) {
#pragma unroll
    for (int s = 0; s < 2; ++s) {
      const int k = kc * 32 + half * 16 + v * 2 + s;
      b[v * 2 + s] = (bf16)sB[(size_t)k * ldc + col0 + n];
    }
  }
  return b;
}

#if USE_TDM
// ---------------------------------------------------------------------------
// Tensor Data Mover: 2D fp32 tile (tile0 contiguous elems x tile1 rows) from
// global (row stride = stride0 elements) into LDS at lds_off. D# bitfields per
// ISA 8.3/8.4: group0 {count=1, lds_addr, global_addr, type=2}; group1
// {data_size=4B code 2, tensor_dim0/1, tile_dim0/1, tensor_dim0_stride}.
// clang-23 builtin takes 6 args: g0, g1, grp2, grp3, extra, cpol.
// ---------------------------------------------------------------------------
__device__ inline void tdm_load_2d_f32(unsigned lds_off, const void* gptr,
                                       unsigned td0, unsigned td1,
                                       unsigned long long stride0,
                                       unsigned tile0, unsigned tile1) {
  const unsigned long long ga = (unsigned long long)(uintptr_t)gptr;
  uint32x4 g0;
  g0[0] = 1u;                                        // count=1 (user descriptor)
  g0[1] = lds_off;                                   // LDS byte address
  g0[2] = (unsigned)ga;                              // global_addr[31:0]
  g0[3] = (unsigned)((ga >> 32) & 0x01FFFFFFull)     // global_addr[56:32]
          | (2u << 30);                              // type=2 ("image")
  int32x8 g1;
  g1[0] = (int)(2u << 16);                           // data_size=4B
  g1[1] = (int)((td0 & 0xFFFFu) << 16);              // tensor_dim0[15:0]
  g1[2] = (int)(((td0 >> 16) & 0xFFFFu) | ((td1 & 0xFFFFu) << 16));
  g1[3] = (int)(((td1 >> 16) & 0xFFFFu) | (tile0 << 16));   // tile_dim0
  g1[4] = (int)(tile1 & 0xFFFFu);                    // tile_dim1 (tile_dim2=0)
  g1[5] = (int)(unsigned)(stride0 & 0xFFFFFFFFull);  // dim0_stride[31:0]
  g1[6] = (int)(unsigned)((stride0 >> 32) & 0xFFFFull);
  g1[7] = 0;
  const int32x4 z4 = {0, 0, 0, 0};
  const int32x8 z8 = {0, 0, 0, 0, 0, 0, 0, 0};
  __builtin_amdgcn_tensor_load_to_lds(g0, g1, z4, z4, z8, 0);
}
__device__ inline unsigned lds_offset_of(const void* p) {
  // generic pointers to LDS carry the LDS byte offset in the low 32 bits
  return (unsigned)(uintptr_t)p;
}
#endif

// ---------------------------------------------------------------------------
// Implicit-GEMM conv. GEMM view [COUT x K] * [K x NP], K = CIN*KH*KW chunked
// by 64 (zero-padded tail). 256 threads = 8 waves as 2 (cout) x 4 (px): block
// tile 32 Cout x 256 px; each wave -> 16 Cout x 64 px = 4 accumulator tiles,
// 8 WMMAs per K-chunk between barriers.
// ---------------------------------------------------------------------------
template <int CIN, int KH, int KW, int STRIDE, int PAD>
__global__ __launch_bounds__(256) void conv_wmma(
    const float* __restrict__ x,     // (N, CIN, HIN, WIN)
    const float* __restrict__ w,     // (COUT, CIN*KH*KW)
    const float* __restrict__ bias,  // (COUT)
    float* __restrict__ y,           // (N, COUT, HOUT, WOUT)
    int N, int HIN, int WIN, int COUT, int HOUT, int WOUT) {
  constexpr int KTOT = CIN * KH * KW;
  const int NP    = N * HOUT * WOUT;
  const int p0    = blockIdx.x * 256;
  const int cout0 = blockIdx.y * 32;
  const int tid   = threadIdx.x;
  const int lane  = tid & 31;
  const int wave  = tid >> 5;
  const int mt    = wave >> 2;  // cout sub-tile: 0..1
  const int jg    = wave & 3;   // px group of 64: 0..3

  __shared__ __align__(16) bf16 sA[32 * 64];   // weights  [m][k]   4 KB
  __shared__ __align__(16) bf16 sB[256 * 64];  // im2col   [px][k] 32 KB

  v8f acc[4] = {};

  for (int kb = 0; kb < KTOT; kb += 64) {
    // ---- stage weights: 512 k-quads, packed bf16x4 -> ds_store_b64 ----
    for (int q = tid; q < 512; q += 256) {
      const int m = q >> 4, kq = (q & 15) * 4;
      const float* wp = w + (size_t)(cout0 + m) * KTOT + kb + kq;
      v4bf pk;
#pragma unroll
      for (int t = 0; t < 4; ++t)
        pk[t] = (bf16)((kb + kq + t < KTOT) ? wp[t] : 0.0f);
      *(v4bf*)(sA + m * 64 + kq) = pk;
    }
    // ---- stage im2col: 4096 k-quads (gather), packed -> ds_store_b64 ----
    for (int q = tid; q < 4096; q += 256) {
      const int px = q >> 4, kq = (q & 15) * 4;
      const int p = p0 + px;
      float vv[4] = {0.0f, 0.0f, 0.0f, 0.0f};
      if (p < NP) {
        const int n  = p / (HOUT * WOUT);
        const int r  = p % (HOUT * WOUT);
        const int oh = r / WOUT, ow = r % WOUT;
#pragma unroll
        for (int t = 0; t < 4; ++t) {
          const int k = kb + kq + t;
          if (k < KTOT) {
            const int cin = k / (KH * KW);
            const int rr  = k % (KH * KW);
            const int kh  = rr / KW, kw = rr % KW;
            const int ih  = oh * STRIDE - PAD + kh;
            const int iw  = ow * STRIDE - PAD + kw;
            if (ih >= 0 && ih < HIN && iw >= 0 && iw < WIN)
              vv[t] = x[((size_t)(n * CIN + cin) * HIN + ih) * WIN + iw];
          }
        }
      }
      v4bf pk;
#pragma unroll
      for (int t = 0; t < 4; ++t) pk[t] = (bf16)vv[t];
      *(v4bf*)(sB + px * 64 + kq) = pk;
    }
    __syncthreads();

    if (kb + 64 < KTOT && lane == 0)  // global_prefetch_b8 for next weights
      __builtin_prefetch(w + (size_t)cout0 * KTOT + kb + 64, 0, 1);

#pragma unroll
    for (int kc = 0; kc < 2; ++kc) {
      const v16bf fa = frag_a_bf16(sA, mt * 16, kc, lane);
#pragma unroll
      for (int sub = 0; sub < 4; ++sub) {
        const v16bf fb = frag_b_bf16(sB, jg * 64 + sub * 16, kc, lane);
        acc[sub] = __builtin_amdgcn_wmma_f32_16x16x32_bf16(
            false, fa, false, fb, (short)0, acc[sub], false, false);
      }
    }
    __syncthreads();
  }

  // ---- epilogue: bias + store fp32 NCHW ----
  const int half = lane >> 4, nidx = lane & 15;
#pragma unroll
  for (int sub = 0; sub < 4; ++sub) {
#pragma unroll
    for (int v = 0; v < 8; ++v) {
      const int cout = cout0 + mt * 16 + v + 8 * half;
      const int p    = p0 + jg * 64 + sub * 16 + nidx;
      if (p < NP) {
        const int n = p / (HOUT * WOUT);
        const int r = p % (HOUT * WOUT);
        y[((size_t)(n * COUT + cout)) * (HOUT * WOUT) + r] =
            acc[sub][v] + bias[cout];
      }
    }
  }
}

// ---------------------------------------------------------------------------
// All-pairs correlation: C[b][i][j] = sum_d f1[d][i]*f2[d][j] / sqrt(D).
// Block tile 32 i x 128 j, K chunks of 64 (D=256 -> 4 chunks). fp32 A/B tiles
// double-buffered in LDS via the Tensor Data Mover (fallback: float4 loads),
// consumed by WMMA bf16. 8 waves as 2 (i) x 4 (j groups of 32).
// ---------------------------------------------------------------------------
__global__ __launch_bounds__(256) void corr_wmma(const float* __restrict__ f,
                                                 float* __restrict__ out,
                                                 int D, int HW) {
  const int b    = blockIdx.z;
  const int i0   = blockIdx.y * 32;
  const int j0   = blockIdx.x * 128;
  const int tid  = threadIdx.x;
  const int lane = tid & 31;
  const int wave = tid >> 5;
  const int mt   = wave >> 2;  // i sub-tile 0..1
  const int jg   = wave & 3;   // j group of 32
  const float scale = rsqrtf((float)D);

  const float* f1 = f + (size_t)b * D * HW;        // fmap1: batches 0..1
  const float* f2 = f + (size_t)(b + 2) * D * HW;  // fmap2: batches 2..3

  __shared__ __align__(16) float tA[2][64 * 32];   // [k][i]  8 KB x2
  __shared__ __align__(16) float tB[2][64 * 128];  // [k][j] 32 KB x2

  const int nchunk = D / 64;

#if USE_TDM
  if (tid == 0) {
    tdm_load_2d_f32(lds_offset_of(&tA[0][0]), f1 + i0, (unsigned)HW,
                    (unsigned)D, (unsigned long long)HW, 32u, 64u);
    tdm_load_2d_f32(lds_offset_of(&tB[0][0]), f2 + j0, (unsigned)HW,
                    (unsigned)D, (unsigned long long)HW, 128u, 64u);
  }
#endif

  v8f acc[2] = {};
  for (int c = 0; c < nchunk; ++c) {
    const int cur = c & 1;
#if USE_TDM
    __builtin_amdgcn_s_wait_tensorcnt(0);  // wave0 gates; others pass
    __syncthreads();                       // tile c visible to all waves
    if (tid == 0 && c + 1 < nchunk) {      // overlap DMA of next chunk
      const size_t koff = (size_t)(c + 1) * 64 * HW;
      tdm_load_2d_f32(lds_offset_of(&tA[(c + 1) & 1][0]), f1 + koff + i0,
                      (unsigned)HW, (unsigned)D, (unsigned long long)HW, 32u,
                      64u);
      tdm_load_2d_f32(lds_offset_of(&tB[(c + 1) & 1][0]), f2 + koff + j0,
                      (unsigned)HW, (unsigned)D, (unsigned long long)HW, 128u,
                      64u);
    }
#else
    // fallback: vectorized float4 staging
    for (int q = tid; q < 512; q += 256) {  // 64x32 / 4
      const int k = q >> 3, iq = (q & 7) * 4;
      *(f32x4*)(&tA[cur][k * 32 + iq]) =
          *(const f32x4*)(f1 + (size_t)(c * 64 + k) * HW + i0 + iq);
    }
    for (int q = tid; q < 2048; q += 256) {  // 64x128 / 4
      const int k = q >> 5, jq = (q & 31) * 4;
      *(f32x4*)(&tB[cur][k * 128 + jq]) =
          *(const f32x4*)(f2 + (size_t)(c * 64 + k) * HW + j0 + jq);
    }
    __syncthreads();
#endif

#pragma unroll
    for (int kc = 0; kc < 2; ++kc) {
      const v16bf fa = frag_a_f32(tA[cur], 32, mt * 16, kc, lane);
#pragma unroll
      for (int sub = 0; sub < 2; ++sub) {
        const v16bf fb = frag_b_f32(tB[cur], 128, jg * 32 + sub * 16, kc, lane);
        acc[sub] = __builtin_amdgcn_wmma_f32_16x16x32_bf16(
            false, fa, false, fb, (short)0, acc[sub], false, false);
      }
    }
    __syncthreads();  // done reading buffer `cur` before it is refilled
  }

  const int half = lane >> 4, nidx = lane & 15;
#pragma unroll
  for (int sub = 0; sub < 2; ++sub) {
#pragma unroll
    for (int v = 0; v < 8; ++v) {
      const int i = i0 + mt * 16 + v + 8 * half;
      const int j = j0 + jg * 32 + sub * 16 + nidx;
      out[((size_t)b * HW + i) * HW + j] = acc[sub][v] * scale;
    }
  }
}

// ---------------------------------------------------------------------------
// InstanceNorm2d (affine=False), elementwise ops, pooling, preprocessing.
// ---------------------------------------------------------------------------
__global__ __launch_bounds__(256) void inorm_stats(const float* __restrict__ x,
                                                   float* __restrict__ stats,
                                                   int HW) {
  const int nc   = blockIdx.x;
  const float* p = x + (size_t)nc * HW;
  float s = 0.0f, s2 = 0.0f;
  for (int i = threadIdx.x; i < HW; i += 256) {
    const float v = p[i];
    s += v;
    s2 += v * v;
  }
  __shared__ float rs[256], rq[256];
  rs[threadIdx.x] = s;
  rq[threadIdx.x] = s2;
  __syncthreads();
  for (int off = 128; off > 0; off >>= 1) {
    if (threadIdx.x < (unsigned)off) {
      rs[threadIdx.x] += rs[threadIdx.x + off];
      rq[threadIdx.x] += rq[threadIdx.x + off];
    }
    __syncthreads();
  }
  if (threadIdx.x == 0) {
    const float m   = rs[0] / (float)HW;
    const float var = rq[0] / (float)HW - m * m;
    stats[2 * nc]     = m;
    stats[2 * nc + 1] = rsqrtf(var + INORM_EPS);
  }
}

__global__ void inorm_apply(float* __restrict__ x,
                            const float* __restrict__ stats, int HW, long total,
                            int relu) {
  const long i = (long)blockIdx.x * blockDim.x + threadIdx.x;
  if (i >= total) return;
  const long nc = i / HW;
  float v = (x[i] - stats[2 * nc]) * stats[2 * nc + 1];
  if (relu) v = fmaxf(v, 0.0f);
  x[i] = v;
}

__global__ void add_relu(const float* __restrict__ a,
                         const float* __restrict__ b, float* __restrict__ o,
                         long n) {
  const long i = (long)blockIdx.x * blockDim.x + threadIdx.x;
  if (i < n) o[i] = fmaxf(a[i] + b[i], 0.0f);
}

__global__ void preprocess(const float* __restrict__ im1,
                           const float* __restrict__ im2,
                           float* __restrict__ out, long per) {
  const long i = (long)blockIdx.x * blockDim.x + threadIdx.x;
  if (i >= 2 * per) return;
  const float v = (i < per) ? im1[i] : im2[i - per];
  out[i] = 2.0f * (v * (1.0f / 255.0f)) - 1.0f;
}

__global__ void avgpool2(const float* __restrict__ in, float* __restrict__ out,
                         int H, int W, long M) {
  const int  OH = H >> 1, OW = W >> 1;
  const long total = M * (long)OH * OW;
  const long i = (long)blockIdx.x * blockDim.x + threadIdx.x;
  if (i >= total) return;
  const long m = i / ((long)OH * OW);
  const int  r = (int)(i % ((long)OH * OW));
  const int  oh = r / OW, ow = r % OW;
  const float* p = in + m * (long)H * W;
  const int base = (2 * oh) * W + 2 * ow;
  out[i] = 0.25f * (p[base] + p[base + 1] + p[base + W] + p[base + W + 1]);
}

// ===========================================================================
// Host orchestration
// ===========================================================================
extern "C" void kernel_launch(void* const* d_in, const int* in_sizes, int n_in,
                              void* d_out, int out_size, void* d_ws,
                              size_t ws_size, hipStream_t stream) {
  (void)in_sizes; (void)n_in; (void)out_size; (void)ws_size;

  // inputs flattened jax-style (sorted dict keys; blocks: b1,b2,(bd,)w1,w2,(wd))
  const float* image1  = (const float*)d_in[0];
  const float* image2  = (const float*)d_in[1];
  const float* conv1_b = (const float*)d_in[2];
  const float* conv1_w = (const float*)d_in[3];
  const float* conv2_b = (const float*)d_in[4];
  const float* conv2_w = (const float*)d_in[5];
  const float* l1b0_b1 = (const float*)d_in[6];
  const float* l1b0_b2 = (const float*)d_in[7];
  const float* l1b0_w1 = (const float*)d_in[8];
  const float* l1b0_w2 = (const float*)d_in[9];
  const float* l1b1_b1 = (const float*)d_in[10];
  const float* l1b1_b2 = (const float*)d_in[11];
  const float* l1b1_w1 = (const float*)d_in[12];
  const float* l1b1_w2 = (const float*)d_in[13];
  const float* l2b0_b1 = (const float*)d_in[14];
  const float* l2b0_b2 = (const float*)d_in[15];
  const float* l2b0_bd = (const float*)d_in[16];
  const float* l2b0_w1 = (const float*)d_in[17];
  const float* l2b0_w2 = (const float*)d_in[18];
  const float* l2b0_wd = (const float*)d_in[19];
  const float* l2b1_b1 = (const float*)d_in[20];
  const float* l2b1_b2 = (const float*)d_in[21];
  const float* l2b1_w1 = (const float*)d_in[22];
  const float* l2b1_w2 = (const float*)d_in[23];
  const float* l3b0_b1 = (const float*)d_in[24];
  const float* l3b0_b2 = (const float*)d_in[25];
  const float* l3b0_bd = (const float*)d_in[26];
  const float* l3b0_w1 = (const float*)d_in[27];
  const float* l3b0_w2 = (const float*)d_in[28];
  const float* l3b0_wd = (const float*)d_in[29];
  const float* l3b1_b1 = (const float*)d_in[30];
  const float* l3b1_b2 = (const float*)d_in[31];
  const float* l3b1_w1 = (const float*)d_in[32];
  const float* l3b1_w2 = (const float*)d_in[33];

  float* out = (float*)d_out;

  // workspace carve (floats)
  const size_t ACT = (size_t)4 * 64 * 192 * 384;  // 18,874,368
  float* B0   = (float*)d_ws;
  float* B1   = B0 + ACT;
  float* B2   = B1 + ACT;
  float* BD   = B2 + ACT;
  float* BIN  = BD + (size_t)4 * 96 * 96 * 192;
  float* STAT = BIN + (size_t)4 * 3 * 384 * 768;

  const int  N = 4;
  const long PER_IMG = (long)2 * 3 * 384 * 768;

  auto grid1d = [](long n) { return (unsigned)((n + 255) / 256); };
  auto cgrid  = [](long np, int cout) {
    return dim3((unsigned)((np + 255) / 256), (unsigned)(cout / 32));
  };
  auto inorm = [&](float* x, int C, int H, int W, int relu) {
    const int HW = H * W;
    inorm_stats<<<N * C, 256, 0, stream>>>(x, STAT, HW);
    const long total = (long)N * C * HW;
    inorm_apply<<<grid1d(total), 256, 0, stream>>>(x, STAT, HW, total, relu);
  };

  // preprocess: normalize + concat -> BIN (4,3,384,768)
  preprocess<<<grid1d(2 * PER_IMG), 256, 0, stream>>>(image1, image2, BIN,
                                                      PER_IMG);

  // stem: 7x7 s2 p3, 3->64 @ 192x384
  {
    const long NP = (long)N * 192 * 384;
    conv_wmma<3, 7, 7, 2, 3><<<cgrid(NP, 64), 256, 0, stream>>>(
        BIN, conv1_w, conv1_b, B0, N, 384, 768, 64, 192, 384);
    inorm(B0, 64, 192, 384, 1);
  }

  // layer1: 2x (64->64, s1) @ 192x384 ; x: B0 -> B0
  {
    const long NP = (long)N * 192 * 384;
    const long T  = (long)N * 64 * 192 * 384;
    conv_wmma<64, 3, 3, 1, 1><<<cgrid(NP, 64), 256, 0, stream>>>(
        B0, l1b0_w1, l1b0_b1, B1, N, 192, 384, 64, 192, 384);
    inorm(B1, 64, 192, 384, 1);
    conv_wmma<64, 3, 3, 1, 1><<<cgrid(NP, 64), 256, 0, stream>>>(
        B1, l1b0_w2, l1b0_b2, B2, N, 192, 384, 64, 192, 384);
    inorm(B2, 64, 192, 384, 1);
    add_relu<<<grid1d(T), 256, 0, stream>>>(B0, B2, B1, T);
    conv_wmma<64, 3, 3, 1, 1><<<cgrid(NP, 64), 256, 0, stream>>>(
        B1, l1b1_w1, l1b1_b1, B0, N, 192, 384, 64, 192, 384);
    inorm(B0, 64, 192, 384, 1);
    conv_wmma<64, 3, 3, 1, 1><<<cgrid(NP, 64), 256, 0, stream>>>(
        B0, l1b1_w2, l1b1_b2, B2, N, 192, 384, 64, 192, 384);
    inorm(B2, 64, 192, 384, 1);
    add_relu<<<grid1d(T), 256, 0, stream>>>(B1, B2, B0, T);
  }

  // layer2: (64->96, s2) + (96->96, s1) @ 96x192 ; x: B0 -> B0
  {
    const long NP = (long)N * 96 * 192;
    const long T  = (long)N * 96 * 96 * 192;
    conv_wmma<64, 3, 3, 2, 1><<<cgrid(NP, 96), 256, 0, stream>>>(
        B0, l2b0_w1, l2b0_b1, B1, N, 192, 384, 96, 96, 192);
    inorm(B1, 96, 96, 192, 1);
    conv_wmma<96, 3, 3, 1, 1><<<cgrid(NP, 96), 256, 0, stream>>>(
        B1, l2b0_w2, l2b0_b2, B2, N, 96, 192, 96, 96, 192);
    inorm(B2, 96, 96, 192, 1);
    conv_wmma<64, 1, 1, 2, 0><<<cgrid(NP, 96), 256, 0, stream>>>(
        B0, l2b0_wd, l2b0_bd, BD, N, 192, 384, 96, 96, 192);
    inorm(BD, 96, 96, 192, 0);
    add_relu<<<grid1d(T), 256, 0, stream>>>(BD, B2, B1, T);
    conv_wmma<96, 3, 3, 1, 1><<<cgrid(NP, 96), 256, 0, stream>>>(
        B1, l2b1_w1, l2b1_b1, B0, N, 96, 192, 96, 96, 192);
    inorm(B0, 96, 96, 192, 1);
    conv_wmma<96, 3, 3, 1, 1><<<cgrid(NP, 96), 256, 0, stream>>>(
        B0, l2b1_w2, l2b1_b2, B2, N, 96, 192, 96, 96, 192);
    inorm(B2, 96, 96, 192, 1);
    add_relu<<<grid1d(T), 256, 0, stream>>>(B1, B2, B0, T);
  }

  // layer3: (96->128, s2) + (128->128, s1) @ 48x96 ; x: B0 -> B0
  {
    const long NP = (long)N * 48 * 96;
    const long T  = (long)N * 128 * 48 * 96;
    conv_wmma<96, 3, 3, 2, 1><<<cgrid(NP, 128), 256, 0, stream>>>(
        B0, l3b0_w1, l3b0_b1, B1, N, 96, 192, 128, 48, 96);
    inorm(B1, 128, 48, 96, 1);
    conv_wmma<128, 3, 3, 1, 1><<<cgrid(NP, 128), 256, 0, stream>>>(
        B1, l3b0_w2, l3b0_b2, B2, N, 48, 96, 128, 48, 96);
    inorm(B2, 128, 48, 96, 1);
    conv_wmma<96, 1, 1, 2, 0><<<cgrid(NP, 128), 256, 0, stream>>>(
        B0, l3b0_wd, l3b0_bd, BD, N, 96, 192, 128, 48, 96);
    inorm(BD, 128, 48, 96, 0);
    add_relu<<<grid1d(T), 256, 0, stream>>>(BD, B2, B1, T);
    conv_wmma<128, 3, 3, 1, 1><<<cgrid(NP, 128), 256, 0, stream>>>(
        B1, l3b1_w1, l3b1_b1, B0, N, 48, 96, 128, 48, 96);
    inorm(B0, 128, 48, 96, 1);
    conv_wmma<128, 3, 3, 1, 1><<<cgrid(NP, 128), 256, 0, stream>>>(
        B0, l3b1_w2, l3b1_b2, B2, N, 48, 96, 128, 48, 96);
    inorm(B2, 128, 48, 96, 1);
    add_relu<<<grid1d(T), 256, 0, stream>>>(B1, B2, B0, T);
  }

  // conv2: 1x1, 128->256 @ 48x96 (raw) -> fmap in B1
  {
    const long NP = (long)N * 48 * 96;
    conv_wmma<128, 1, 1, 1, 0><<<cgrid(NP, 256), 256, 0, stream>>>(
        B0, conv2_w, conv2_b, B1, N, 48, 96, 256, 48, 96);
  }

  // all-pairs correlation -> pyramid level 0
  const int HW = 48 * 96;  // 4608
  corr_wmma<<<dim3(HW / 128, HW / 32, 2), 256, 0, stream>>>(B1, out, 256, HW);

  // pyramid: three 2x2 avg-pools over (9216, h, w)
  const long M  = (long)2 * HW;  // 9216
  const long L1 = M * 48 * 96;
  const long L2 = L1 + M * 24 * 48;
  const long L3 = L2 + M * 12 * 24;
  avgpool2<<<grid1d(M * 24 * 48), 256, 0, stream>>>(out, out + L1, 48, 96, M);
  avgpool2<<<grid1d(M * 12 * 24), 256, 0, stream>>>(out + L1, out + L2, 24, 48, M);
  avgpool2<<<grid1d(M * 6 * 12), 256, 0, stream>>>(out + L2, out + L3, 12, 24, M);
}
